// FiloPrioriV10Light_40355512713952
// MI455X (gfx1250) — matrix operand order repaired and checked by reading, not compiled, not run on hardware
//
#include <hip/hip_runtime.h>

#define DEV __device__ __forceinline__

typedef __bf16 bf16_t;
typedef __attribute__((ext_vector_type(16))) __bf16 v16bf;
typedef __attribute__((ext_vector_type(8)))  float  v8f;
typedef int v4i __attribute__((vector_size(16)));                 // matches builtin param
typedef __attribute__((address_space(1))) v4i* g_v4i_ptr;         // global int4*
typedef __attribute__((address_space(3))) v4i* l_v4i_ptr;         // LDS int4*

// CDNA5 async global->LDS path (ASYNCcnt). Guarded so the file compiles on
// toolchains that do not expose the builtin (falls back to plain LDS copy).
#if defined(__has_builtin)
# if __has_builtin(__builtin_amdgcn_global_load_async_to_lds_b128)
#  define HAVE_ASYNC_LDS 1
# endif
# if __has_builtin(__builtin_amdgcn_s_wait_asynccnt)
#  define WAIT_ASYNC(n) __builtin_amdgcn_s_wait_asynccnt(n)
# endif
#endif
#ifndef HAVE_ASYNC_LDS
# define HAVE_ASYNC_LDS 0
#endif
#ifndef WAIT_ASYNC
# define WAIT_ASYNC(n) asm volatile("s_wait_asynccnt %0" :: "i"(n) : "memory")
#endif

// ---- problem constants (match reference) ----
static constexpr int BN   = 4096;     // batch
static constexpr int NN   = 10000;    // nodes
static constexpr int EN   = 160000;   // edges
static constexpr int DIN  = 768;
static constexpr int HID  = 256;
static constexpr int HC   = 1024;     // H*C
static constexpr int HEU  = 6;

DEV float gelu_f(float x) { return 0.5f * x * (1.0f + erff(x * 0.70710678118654752f)); }
DEV unsigned fkey(float f) {
  unsigned s = __float_as_uint(f);
  return (s & 0x80000000u) ? ~s : (s | 0x80000000u);   // monotonic float->uint
}
DEV float unfkey(unsigned k) {
  return __uint_as_float((k & 0x80000000u) ? (k ^ 0x80000000u) : ~k);
}

// =====================================================================
// Pack fp32 row-major [M,K] (optionally row-gathered via idx) into the
// CDNA5 WMMA bf16 A-matrix per-lane layout (16x32 tiles):
//   lane 0-15 : M=lane,    halves 0-7 = K 0..7,  halves 8-15 = K 16..23
//   lane16-31 : M=lane-16, halves 0-7 = K 8..15, halves 8-15 = K 24..31
// Tile slot index = ((mt*Kt + kt)*32 + lane)*16 + j
// =====================================================================
__global__ void pack_a_kernel(const float* __restrict__ src, bf16_t* __restrict__ dst,
                              int Mt, int Kt, int ld, const int* __restrict__ idx) {
  long t = (long)blockIdx.x * blockDim.x + threadIdx.x;
  long total = (long)Mt * Kt * 512;
  if (t >= total) return;
  int j    = (int)(t & 15);
  int lane = (int)((t >> 4) & 31);
  long tile = t >> 9;
  int kt = (int)(tile % Kt);
  int mt = (int)(tile / Kt);
  int row = mt * 16 + (lane & 15);
  int kbase = (lane < 16) ? 0 : 8;
  int k = kt * 32 + ((j < 8) ? (kbase + j) : (kbase + 8 + j));
  int r = idx ? idx[row] : row;
  dst[t] = (bf16_t)src[(long)r * ld + k];
}

// =====================================================================
// Pack weight W [Nw,K] row-major (computing x @ W^T, so B[k,n] = W[n,k])
// into WMMA bf16 B-matrix layout (32x16 tiles):
//   lane 0-15 : N=lane,    halves 0-7 = K 0..7,  halves 8-15 = K 16..23
//   lane16-31 : N=lane-16, halves 0-7 = K 8..15, halves 8-15 = K 24..31
// =====================================================================
__global__ void pack_b_kernel(const float* __restrict__ w, bf16_t* __restrict__ dst,
                              int Nt, int Kt, int K) {
  long t = (long)blockIdx.x * blockDim.x + threadIdx.x;
  long total = (long)Nt * Kt * 512;
  if (t >= total) return;
  int j    = (int)(t & 15);
  int lane = (int)((t >> 4) & 31);
  long tile = t >> 9;
  int kt = (int)(tile % Kt);
  int nt = (int)(tile / Kt);
  int n = nt * 16 + (lane & 15);
  int kbase = (lane < 16) ? 0 : 8;
  int k = kt * 32 + kbase + ((j < 8) ? j : (j + 8));
  dst[t] = (bf16_t)w[(long)n * K + k];
}

// =====================================================================
// WMMA GEMM: C = act(A@B + bias).
// Block = 256 threads = 8 waves. Block tile = 128 rows (8 M-tiles, one per
// wave) x 32 cols (2 N-tiles). B tiles (2 x 1KB per k-step) are staged in
// LDS with GLOBAL_LOAD_ASYNC_TO_LDS_B128 (ASYNCcnt double-buffer pipeline,
// waves 0-3 stage 512B each) and read back with ds_load_b128; A tiles come
// straight from global (per-wave, pipelined by the hardware + prefetch).
// act: 0=none, 1=gelu, 2=sigmoid
// =====================================================================
__global__ void wmma_gemm_kernel(const bf16_t* __restrict__ pa, const bf16_t* __restrict__ pb,
                                 float* __restrict__ c, const float* __restrict__ bias,
                                 int Mt, int Nt, int Kt, int ldc, int act) {
  __shared__ __align__(32) bf16_t Bs[2][2][512];   // [buf][n-tile][1KB tile]

  const int tid  = threadIdx.x;
  const int wave = tid >> 5;
  const int lane = tid & 31;
  const int Ng   = Nt >> 1;
  const int mg   = blockIdx.x / Ng;
  const int ntp  = blockIdx.x % Ng;
  const int mt   = mg * 8 + wave;
  const bool active = (mt < Mt);     // wave-uniform: EXEC stays all-ones

  const char* bsrc = (const char*)pb;

  auto stage = [&](int buf, int kt) {
    if (tid < 128) {                                   // waves 0-3 stage 2KB
      int tile = tid >> 6;                             // 0 or 1
      int off  = (tid & 63) * 16;                      // byte offset in tile
      const char* g = bsrc + ((size_t)((2 * ntp + tile) * Kt + kt) << 10) + off;
      bf16_t* l = &Bs[buf][tile][0] + (off >> 1);
#if HAVE_ASYNC_LDS
      __builtin_amdgcn_global_load_async_to_lds_b128(
          (g_v4i_ptr)(v4i*)(void*)g, (l_v4i_ptr)(v4i*)l, 0, 0);
#else
      *(v4i*)l = *(const v4i*)g;
#endif
    }
  };

  const v16bf* Av = (const v16bf*)pa;
  const long abase = (long)mt * Kt * 32 + lane;

  v8f acc0 = {}, acc1 = {};
  stage(0, 0);
  for (int kt = 0; kt < Kt; ++kt) {
    const int cur = kt & 1;
    if (kt + 1 < Kt) stage(cur ^ 1, kt + 1);
#if HAVE_ASYNC_LDS
    if (wave < 4) {                                    // wave-uniform waits
      if (kt + 1 < Kt) WAIT_ASYNC(1);
      else             WAIT_ASYNC(0);
    }
#endif
    __syncthreads();
    if (active) {
      v16bf a = Av[abase + (long)kt * 32];
      if (kt + 1 < Kt)
        __builtin_prefetch((const void*)&Av[abase + (long)(kt + 1) * 32], 0, 3);
      v16bf b0 = *(const v16bf*)&Bs[cur][0][lane * 16];
      v16bf b1 = *(const v16bf*)&Bs[cur][1][lane * 16];
      acc0 = __builtin_amdgcn_wmma_f32_16x16x32_bf16(false, a, false, b0,
                                                     (short)0, acc0, false, false);
      acc1 = __builtin_amdgcn_wmma_f32_16x16x32_bf16(false, a, false, b1,
                                                     (short)0, acc1, false, false);
    }
    __syncthreads();                                   // protect buffer reuse
  }

  if (active) {
    // C/D layout: VGPR v -> M = v (lanes 0-15) or v+8 (lanes 16-31); N = lane&15
    int rbase = mt * 16 + ((lane < 16) ? 0 : 8);
    int col0  = (2 * ntp) * 16 + (lane & 15);
    int col1  = col0 + 16;
    float bv0 = bias ? bias[col0] : 0.0f;
    float bv1 = bias ? bias[col1] : 0.0f;
#pragma unroll
    for (int v = 0; v < 8; ++v) {
      float x0 = acc0[v] + bv0;
      float x1 = acc1[v] + bv1;
      if (act == 1)      { x0 = gelu_f(x0); x1 = gelu_f(x1); }
      else if (act == 2) { x0 = 1.0f / (1.0f + expf(-x0));
                           x1 = 1.0f / (1.0f + expf(-x1)); }
      long r = (long)(rbase + v) * ldc;
      c[r + col0] = x0;
      c[r + col1] = x1;
    }
  }
}

// ---------------- init / reductions ----------------
__global__ void zero_f4_kernel(float4* __restrict__ p, long n4) {
  long t = (long)blockIdx.x * blockDim.x + threadIdx.x;
  if (t < n4) p[t] = make_float4(0.f, 0.f, 0.f, 0.f);
}

__global__ void init_small_kernel(float* __restrict__ meanAcc, float* __restrict__ denom,
                                  unsigned* __restrict__ amaxKey, int nh) {
  int t = blockIdx.x * blockDim.x + threadIdx.x;
  if (t == 0) meanAcc[0] = 0.f;
  if (t < nh) { denom[t] = 0.f; amaxKey[t] = fkey(-3.402823466e38f); }
}

__global__ void mean_reduce_kernel(const float* __restrict__ ew, float* __restrict__ accum, int E) {
  __shared__ float sm[256];
  int t = blockIdx.x * 256 + threadIdx.x;
  sm[threadIdx.x] = (t < E) ? ew[t] : 0.f;
  __syncthreads();
  for (int s = 128; s > 0; s >>= 1) {
    if (threadIdx.x < s) sm[threadIdx.x] += sm[threadIdx.x + s];
    __syncthreads();
  }
  if (threadIdx.x == 0) atomicAdd(accum, sm[0]);
}

// k_edge[h] = sum_c lin_edge_w[h*256+c] * att_edge[h*256+c]
__global__ void kedge_kernel(const float* __restrict__ lew, const float* __restrict__ ae,
                             float* __restrict__ kedge) {
  int h = threadIdx.x;
  if (h >= 4) return;
  float s = 0.f;
  for (int c = 0; c < 256; ++c) s += lew[h * 256 + c] * ae[h * 256 + c];
  kedge[h] = s;
}

// a_src[n,h], a_dst[n,h] = sum_c x_proj[n, h*256+c] * att_{src,dst}[h,c]
__global__ void attdot_kernel(const float* __restrict__ xproj, const float* __restrict__ asrc_w,
                              const float* __restrict__ adst_w, float* __restrict__ asrc,
                              float* __restrict__ adst, int N) {
  int t = blockIdx.x * blockDim.x + threadIdx.x;
  if (t >= N * 4) return;
  int n = t >> 2, h = t & 3;
  const float* xp = xproj + (long)n * HC + h * 256;
  float s0 = 0.f, s1 = 0.f;
  for (int c = 0; c < 256; ++c) {
    float x = xp[c];
    s0 += x * asrc_w[h * 256 + c];
    s1 += x * adst_w[h * 256 + c];
  }
  asrc[t] = s0; adst[t] = s1;
}

// ---------------- GAT edge phase ----------------
__global__ void gat_alpha_kernel(const int* __restrict__ ei, const float* __restrict__ ew,
                                 const float* __restrict__ meanAcc, const float* __restrict__ kedge,
                                 const float* __restrict__ asrc, const float* __restrict__ adst,
                                 float* __restrict__ alpha, unsigned* __restrict__ amaxKey,
                                 int E, int N) {
  int t = blockIdx.x * blockDim.x + threadIdx.x;
  int ET = E + N;
  if (t >= ET * 4) return;
  int e = t >> 2, h = t & 3;
  int s, d; float w;
  if (e < E) { s = ei[e]; d = ei[E + e]; w = ew[e]; }
  else       { s = d = e - E; w = meanAcc[0] / (float)E; }
  float a = asrc[s * 4 + h] + adst[d * 4 + h] + w * kedge[h];
  a = (a > 0.f) ? a : 0.2f * a;           // leaky_relu(0.2)
  alpha[t] = a;
  atomicMax(&amaxKey[d * 4 + h], fkey(a));
}

__global__ void gat_exp_kernel(const int* __restrict__ ei, float* __restrict__ alpha,
                               const unsigned* __restrict__ amaxKey, float* __restrict__ denom,
                               int E, int N) {
  int t = blockIdx.x * blockDim.x + threadIdx.x;
  int ET = E + N;
  if (t >= ET * 4) return;
  int e = t >> 2, h = t & 3;
  int d = (e < E) ? ei[E + e] : (e - E);
  float ex = expf(alpha[t] - unfkey(amaxKey[d * 4 + h]));
  alpha[t] = ex;
  atomicAdd(&denom[d * 4 + h], ex);
}

// scatter: out[dst, :] += coef * x_proj[src, :]; one thread = one edge x 4 floats
__global__ void gat_scatter_kernel(const int* __restrict__ ei, const float* __restrict__ xproj,
                                   const float* __restrict__ expa, const float* __restrict__ denom,
                                   float* __restrict__ out, int E, int N) {
  long t = (long)blockIdx.x * blockDim.x + threadIdx.x;
  long total = (long)(E + N) * 256;
  if (t >= total) return;
  int chunk = (int)(t & 255);
  long e = t >> 8;
  int s, d;
  if (e < E) { s = ei[e]; d = ei[E + e]; }
  else       { s = d = (int)(e - E); }
  int h = chunk >> 6;
  float coef = expa[e * 4 + h] / denom[d * 4 + h];
  int c0 = chunk * 4;
  const float4 xv = *(const float4*)(xproj + (long)s * HC + c0);
  float* o = out + (long)d * HC + c0;
  atomicAdd(o + 0, xv.x * coef);
  atomicAdd(o + 1, xv.y * coef);
  atomicAdd(o + 2, xv.z * coef);
  atomicAdd(o + 3, xv.w * coef);
}

__global__ void elu_bias_kernel(float* __restrict__ out, const float* __restrict__ bias, long n) {
  long t = (long)blockIdx.x * blockDim.x + threadIdx.x;
  if (t >= n) return;
  float x = out[t] + bias[t & (HC - 1)];
  out[t] = (x > 0.f) ? x : expm1f(x);
}

// ---------------- fusion tail ----------------
__global__ void heur_kernel(const float* __restrict__ hf, const float* __restrict__ w,
                            const float* __restrict__ b, float* __restrict__ out) {
  int t = blockIdx.x * blockDim.x + threadIdx.x;
  if (t >= BN * HID) return;
  int bb = t >> 8, j = t & 255;
  float s = b[j];
#pragma unroll
  for (int i = 0; i < HEU; ++i) s += hf[bb * HEU + i] * w[j * HEU + i];
  out[t] = gelu_f(s);
}

__global__ void fuse_kernel(const float* __restrict__ ggate, const float* __restrict__ hcat,
                            const float* __restrict__ hheur, const float* __restrict__ hw,
                            float* __restrict__ hfused) {
  int t = blockIdx.x * blockDim.x + threadIdx.x;
  if (t >= BN * HID) return;
  int bb = t >> 8, j = t & 255;
  float g  = ggate[t];
  float hs = hcat[bb * 512 + j];
  float ht = hcat[bb * 512 + 256 + j];
  float hl = g * hs + (1.f - g) * ht;
  float w  = hw[0];
  hfused[t] = w * hheur[t] + (1.f - w) * hl;
}

__global__ void score_kernel(const float* __restrict__ hs1, const float* __restrict__ sw2,
                             const float* __restrict__ sb2, float* __restrict__ out) {
  int b = blockIdx.x * blockDim.x + threadIdx.x;
  if (b >= BN) return;
  float s = sb2[0];
  for (int j = 0; j < HID; ++j) s += hs1[b * HID + j] * sw2[j];
  out[b] = s;
}

// =====================================================================
// launch
// =====================================================================
static inline int cdiv(long a, long b) { return (int)((a + b - 1) / b); }

extern "C" void kernel_launch(void* const* d_in, const int* in_sizes, int n_in,
                              void* d_out, int out_size, void* d_ws, size_t ws_size,
                              hipStream_t stream) {
  const float* sem_emb  = (const float*)d_in[0];
  const float* graph_x  = (const float*)d_in[1];
  const int*   ei       = (const int*)  d_in[2];
  const float* ew       = (const float*)d_in[3];
  const float* heur_f   = (const float*)d_in[4];
  const int*   node_idx = (const int*)  d_in[5];
  const float* sem_w1   = (const float*)d_in[6];
  const float* sem_b1   = (const float*)d_in[7];
  const float* sem_w2   = (const float*)d_in[8];
  const float* sem_b2   = (const float*)d_in[9];
  const float* lin_w    = (const float*)d_in[10];
  const float* lin_ew   = (const float*)d_in[11];
  const float* att_src  = (const float*)d_in[12];
  const float* att_dst  = (const float*)d_in[13];
  const float* att_edge = (const float*)d_in[14];
  const float* gat_bias = (const float*)d_in[15];
  const float* struct_w = (const float*)d_in[16];
  const float* struct_b = (const float*)d_in[17];
  const float* gate_w   = (const float*)d_in[18];
  const float* gate_b   = (const float*)d_in[19];
  const float* heur_w   = (const float*)d_in[20];
  const float* heur_b   = (const float*)d_in[21];
  const float* hw       = (const float*)d_in[22];
  const float* sw1      = (const float*)d_in[23];
  const float* sb1      = (const float*)d_in[24];
  const float* sw2      = (const float*)d_in[25];
  const float* sb2      = (const float*)d_in[26];
  float* scores = (float*)d_out;

  char* ws = (char*)d_ws;
  constexpr size_t MB = 1u << 20;
  bf16_t*  packA   = (bf16_t*)(ws + 0);           // 16 MB, reused per GEMM
  bf16_t*  packB   = (bf16_t*)(ws + 16 * MB);     //  2 MB, reused per GEMM
  float*   h1      = (float*) (ws + 18 * MB);     //  4 MB  [B,256]
  float*   xproj   = (float*) (ws + 22 * MB);     // 40 MB  [N,1024]
  float*   gout    = (float*) (ws + 62 * MB);     // 40 MB  [N,1024] accum -> h_graph
  float*   alpha   = (float*) (ws + 102 * MB);    //  3 MB  [(E+N)*4]
  unsigned* amaxK  = (unsigned*)(ws + 105 * MB);  // 640 KB
  float*   denom   = (float*) (ws + 105 * MB + 262144);
  float*   asrc    = (float*) (ws + 105 * MB + 524288);
  float*   adst    = (float*) (ws + 105 * MB + 786432);
  float*   kedge   = (float*) (ws + 105 * MB + 1048576);   // 4 floats
  float*   meanAcc = (float*) (ws + 105 * MB + 1048576 + 256);
  float*   hcat    = (float*) (ws + 107 * MB);    //  8 MB  [B,512] = [h_sem | h_struct]
  float*   ggate   = (float*) (ws + 115 * MB);    //  4 MB
  float*   hheur   = (float*) (ws + 119 * MB);    //  4 MB
  float*   hfused  = (float*) (ws + 123 * MB);    //  4 MB
  float*   hs1     = (float*) (ws + 127 * MB);    //  4 MB

  const int TB = 256;
  const int ET = EN + NN;

  auto gemm = [&](const bf16_t* A, const bf16_t* Bp, float* C, const float* bias,
                  int Mt, int Nt, int Kt, int ldc, int act) {
    int blocks = cdiv(Mt, 8) * (Nt >> 1);
    wmma_gemm_kernel<<<blocks, 256, 0, stream>>>(A, Bp, C, bias, Mt, Nt, Kt, ldc, act);
  };

  // ---- init accumulators (every call; deterministic) ----
  init_small_kernel<<<cdiv(NN * 4, TB), TB, 0, stream>>>(meanAcc, denom, amaxK, NN * 4);
  zero_f4_kernel<<<cdiv((long)NN * HC / 4, TB), TB, 0, stream>>>((float4*)gout, (long)NN * HC / 4);
  mean_reduce_kernel<<<cdiv(EN, TB), TB, 0, stream>>>(ew, meanAcc, EN);
  kedge_kernel<<<1, 32, 0, stream>>>(lin_ew, att_edge, kedge);

  // ---- semantic layer 1: [B,1536] @ sem_w1^T + b -> GELU -> h1 [B,256] ----
  { int Mt = BN / 16, Nt = HID / 16, Kt = 1536 / 32;
    pack_b_kernel<<<cdiv((long)Nt * Kt * 512, TB), TB, 0, stream>>>(sem_w1, packB, Nt, Kt, 1536);
    pack_a_kernel<<<cdiv((long)Mt * Kt * 512, TB), TB, 0, stream>>>(sem_emb, packA, Mt, Kt, 1536, nullptr);
    gemm(packA, packB, h1, sem_b1, Mt, Nt, Kt, HID, 1);
  }
  // ---- semantic layer 2: h1 @ sem_w2^T + b -> h_sem (hcat cols 0..255) ----
  { int Mt = BN / 16, Nt = HID / 16, Kt = HID / 32;
    pack_b_kernel<<<cdiv((long)Nt * Kt * 512, TB), TB, 0, stream>>>(sem_w2, packB, Nt, Kt, HID);
    pack_a_kernel<<<cdiv((long)Mt * Kt * 512, TB), TB, 0, stream>>>(h1, packA, Mt, Kt, HID, nullptr);
    gemm(packA, packB, hcat, sem_b2, Mt, Nt, Kt, 512, 0);
  }
  // ---- GAT projection: graph_x [N,768] @ lin_w^T -> x_proj [N,1024] ----
  { int Mt = NN / 16, Nt = HC / 16, Kt = DIN / 32;
    pack_b_kernel<<<cdiv((long)Nt * Kt * 512, TB), TB, 0, stream>>>(lin_w, packB, Nt, Kt, DIN);
    pack_a_kernel<<<cdiv((long)Mt * Kt * 512, TB), TB, 0, stream>>>(graph_x, packA, Mt, Kt, DIN, nullptr);
    gemm(packA, packB, xproj, nullptr, Mt, Nt, Kt, HC, 0);
  }
  // ---- attention dots, alpha, softmax, scatter ----
  attdot_kernel<<<cdiv(NN * 4, TB), TB, 0, stream>>>(xproj, att_src, att_dst, asrc, adst, NN);
  gat_alpha_kernel<<<cdiv((long)ET * 4, TB), TB, 0, stream>>>(ei, ew, meanAcc, kedge, asrc, adst, alpha, amaxK, EN, NN);
  gat_exp_kernel<<<cdiv((long)ET * 4, TB), TB, 0, stream>>>(ei, alpha, amaxK, denom, EN, NN);
  gat_scatter_kernel<<<cdiv((long)ET * 256, TB), TB, 0, stream>>>(ei, xproj, alpha, denom, gout, EN, NN);
  elu_bias_kernel<<<cdiv((long)NN * HC, TB), TB, 0, stream>>>(gout, gat_bias, (long)NN * HC);

  // ---- structural proj: gather rows + GEMM + GELU -> hcat cols 256..511 ----
  { int Mt = BN / 16, Nt = HID / 16, Kt = HC / 32;
    pack_b_kernel<<<cdiv((long)Nt * Kt * 512, TB), TB, 0, stream>>>(struct_w, packB, Nt, Kt, HC);
    pack_a_kernel<<<cdiv((long)Mt * Kt * 512, TB), TB, 0, stream>>>(gout, packA, Mt, Kt, HC, node_idx);
    gemm(packA, packB, hcat + 256, struct_b, Mt, Nt, Kt, 512, 1);
  }
  // ---- heuristic proj (K=6, scalar) ----
  heur_kernel<<<cdiv(BN * HID, TB), TB, 0, stream>>>(heur_f, heur_w, heur_b, hheur);

  // ---- gate: sigmoid(hcat [B,512] @ gate_w^T + b) ----
  { int Mt = BN / 16, Nt = HID / 16, Kt = 512 / 32;
    pack_b_kernel<<<cdiv((long)Nt * Kt * 512, TB), TB, 0, stream>>>(gate_w, packB, Nt, Kt, 512);
    pack_a_kernel<<<cdiv((long)Mt * Kt * 512, TB), TB, 0, stream>>>(hcat, packA, Mt, Kt, 512, nullptr);
    gemm(packA, packB, ggate, gate_b, Mt, Nt, Kt, HID, 2);
  }
  // ---- residual fusion ----
  fuse_kernel<<<cdiv(BN * HID, TB), TB, 0, stream>>>(ggate, hcat, hheur, hw, hfused);

  // ---- scorer layer 1: gelu(hfused @ sw1^T + b) ----
  { int Mt = BN / 16, Nt = HID / 16, Kt = HID / 32;
    pack_b_kernel<<<cdiv((long)Nt * Kt * 512, TB), TB, 0, stream>>>(sw1, packB, Nt, Kt, HID);
    pack_a_kernel<<<cdiv((long)Mt * Kt * 512, TB), TB, 0, stream>>>(hfused, packA, Mt, Kt, HID, nullptr);
    gemm(packA, packB, hs1, sb1, Mt, Nt, Kt, HID, 1);
  }
  // ---- final dot ----
  score_kernel<<<cdiv(BN, TB), TB, 0, stream>>>(hs1, sw2, sb2, scores);

  (void)in_sizes; (void)n_in; (void)out_size; (void)ws_size;
}
